// DRMM_20461224198560
// MI455X (gfx1250) — matrix operand order repaired and checked by reading, not compiled, not run on hardware
//
#include <hip/hip_runtime.h>
#include <hip/hip_bf16.h>
#include <math.h>

// ---------------------------------------------------------------------------
// DRMM fused pipeline for MI455X (gfx1250, wave32, WMMA bf16).
// B=64, Q=32, D=4096, E=300, NBINS=30.
// ---------------------------------------------------------------------------

#define B_SZ   64
#define Q_SZ   32
#define D_SZ   4096
#define E_SZ   300
#define NBINS  30
#define DT     64            // doc rows per block
#define NDT    (D_SZ / DT)   // 64 doc tiles per batch
#define KPAD   320           // K rounded up to multiple of 32
#define QSTR   328           // padded row stride (bf16): 164 dwords -> conflict-free b128

typedef __bf16 bf16;
typedef __attribute__((ext_vector_type(16))) __bf16 v16bf;
typedef __attribute__((ext_vector_type(8)))  __bf16 v8bf;
typedef __attribute__((ext_vector_type(8)))  float   v8f;

union FragBF { v16bf v; v8bf h[2]; };

// ---------------------------------------------------------------------------
// Kernel 1: normalize query rows -> bf16 padded rows in ws; zero global hist.
// grid = B*Q blocks of 32 threads (one wave per row).
// ---------------------------------------------------------------------------
__global__ __launch_bounds__(32)
void drmm_qnorm(const float* __restrict__ q, bf16* __restrict__ qn,
                float* __restrict__ hist)
{
    const int row = blockIdx.x;          // b*Q + qidx
    const int ln  = threadIdx.x;         // 0..31
    const float* src = q + (size_t)row * E_SZ;

    float vals[10];
    float ss = 0.f;
#pragma unroll
    for (int i = 0; i < 10; ++i) {
        int k = ln + 32 * i;
        float x = (k < E_SZ) ? src[k] : 0.f;
        vals[i] = x;
        ss += x * x;
    }
#pragma unroll
    for (int o = 16; o; o >>= 1) ss += __shfl_xor(ss, o);
    const float rn = rsqrtf(ss);

    bf16* dst = qn + (size_t)row * QSTR;
#pragma unroll
    for (int i = 0; i < 11; ++i) {
        int k = ln + 32 * i;
        if (k < QSTR) {
            float v = (i < 10 && k < E_SZ) ? vals[i] * rn : 0.f;
            dst[k] = (bf16)v;
        }
    }
    if (ln < NBINS) hist[(size_t)row * NBINS + ln] = 0.f;
}

// ---------------------------------------------------------------------------
// Kernel 2: fused (normalize docs -> bf16 LDS tile) x (WMMA interaction)
//           x (histogram). One block per (batch, 64-doc-row tile).
// 128 threads = 4 waves; wave w owns doc rows [w*16, w*16+16), all 32 q rows.
// ---------------------------------------------------------------------------
__global__ __launch_bounds__(128)
void drmm_interact_hist(const float* __restrict__ doc,
                        const bf16*  __restrict__ qn,
                        float*       __restrict__ hist)
{
    __shared__ bf16     dtile[DT * QSTR];     // 41,984 B
    __shared__ float    rnorm[DT];            //    256 B
    __shared__ unsigned lhist[Q_SZ * NBINS];  //  3,840 B   (total < 64 KB)

    const int tid   = threadIdx.x;
    const int wv    = tid >> 5;
    const int ln    = tid & 31;
    const int b     = blockIdx.x >> 6;        // / NDT
    const int dt    = blockIdx.x & (NDT - 1);
    const int dbase = dt * DT;

    for (int i = tid; i < Q_SZ * NBINS; i += 128) lhist[i] = 0u;

    // ---- doc row norms: 2 threads per row, contiguous 150-float streams ----
    {
        const int r = tid >> 1, half = tid & 1;
        const float* rowp = doc + ((size_t)b * D_SZ + dbase + r) * E_SZ + half * 150;
        float ss = 0.f;
        for (int k = 0; k < 150; ++k) { float x = rowp[k]; ss += x * x; }
        ss += __shfl_xor(ss, 1);              // partner lane, wave32
        if (half == 0) rnorm[r] = rsqrtf(ss);
    }
    __syncthreads();

    // ---- normalize + convert to bf16 LDS tile (reads hit L2), zero-pad K ----
    for (int r = wv; r < DT; r += 4) {
        const float rn = rnorm[r];
        const float* rowp = doc + ((size_t)b * D_SZ + dbase + r) * E_SZ;
        for (int k = ln; k < QSTR; k += 32) {
            float v = (k < E_SZ) ? rowp[k] * rn : 0.f;
            dtile[r * QSTR + k] = (bf16)v;
        }
    }
    __syncthreads();

    // ---- WMMA K-loop:  C[32 q x 16 d] per wave, K = 320 (zero padded) ------
    const int r15 = ln & 15;                  // M (A) / N (B) index within tile
    const int kh  = ln >> 4;                  // K half-wave selector
    const int drow = wv * 16 + r15;

    const bf16* qbase = qn + ((size_t)b * Q_SZ) * QSTR;

    v8f c0{};   // q rows 0..15
    v8f c1{};   // q rows 16..31
#pragma unroll
    for (int K0 = 0; K0 < KPAD; K0 += 32) {
        FragBF a0, a1, bm;
        // A frag: lanes 0-15 K {0..7,16..23}, lanes 16-31 K {8..15,24..31}
        const v8bf* pa0 = (const v8bf*)(qbase + (size_t)r15 * QSTR + K0 + 8 * kh);
        a0.h[0] = pa0[0];  a0.h[1] = pa0[2];
        const v8bf* pa1 = (const v8bf*)(qbase + (size_t)(16 + r15) * QSTR + K0 + 8 * kh);
        a1.h[0] = pa1[0];  a1.h[1] = pa1[2];
        // B frag: lane = column (doc row), lanes 0-15 K {0..7,16..23}, lanes 16-31 K {8..15,24..31}
        const v8bf* pb = (const v8bf*)(&dtile[drow * QSTR + K0 + 8 * kh]);
        bm.h[0] = pb[0];   bm.h[1] = pb[2];

        c0 = __builtin_amdgcn_wmma_f32_16x16x32_bf16(false, a0.v, false, bm.v,
                                                     (short)0, c0, false, false);
        c1 = __builtin_amdgcn_wmma_f32_16x16x32_bf16(false, a1.v, false, bm.v,
                                                     (short)0, c1, false, false);
    }

    // ---- bin the 16 cosine values held by this lane into the LDS hist ------
#pragma unroll
    for (int v = 0; v < 8; ++v) {
        const int m = v + 8 * kh;             // q row within 16-row tile
        float x0 = c0[v];
        float x1 = c1[v];
        int b0 = (int)floorf((x0 + 1.f) * (NBINS * 0.5f));
        int b1 = (int)floorf((x1 + 1.f) * (NBINS * 0.5f));
        b0 = min(max(b0, 0), NBINS - 1);
        b1 = min(max(b1, 0), NBINS - 1);
        atomicAdd(&lhist[m * NBINS + b0], 1u);
        atomicAdd(&lhist[(16 + m) * NBINS + b1], 1u);
    }
    __syncthreads();

    // ---- flush block histogram (integer counts in f32 -> exact, determ.) ---
    for (int i = tid; i < Q_SZ * NBINS; i += 128) {
        float v = (float)lhist[i];
        if (v != 0.f)
            atomicAdd(&hist[(size_t)b * Q_SZ * NBINS + i], v);
    }
}

// ---------------------------------------------------------------------------
// Kernel 3: gate softmax + masked log1p(hist) -> tanh MLP -> gated sum.
// One wave per batch; lane = query row.
// ---------------------------------------------------------------------------
__global__ __launch_bounds__(32)
void drmm_final(const float* __restrict__ q, const int* __restrict__ qlen,
                const float* __restrict__ hist,
                const float* __restrict__ W1, const float* __restrict__ b1,
                const float* __restrict__ W2, const float* __restrict__ b2,
                const float* __restrict__ W3, const float* __restrict__ b3,
                const float* __restrict__ Wg, const float* __restrict__ bg,
                float* __restrict__ out)
{
    __shared__ float wg_s[E_SZ];
    const int b  = blockIdx.x;
    const int ln = threadIdx.x;               // query row
    for (int k = ln; k < E_SZ; k += 32) wg_s[k] = Wg[k];
    __syncthreads();

    // gate logit on NORMALIZED query row (fp32 recompute for accuracy)
    const float* qrow = q + ((size_t)b * Q_SZ + ln) * E_SZ;
    float ss = 0.f, dot = 0.f;
    for (int k = 0; k < E_SZ; ++k) {
        float x = qrow[k];
        ss  += x * x;
        dot += x * wg_s[k];
    }
    float logit = dot * rsqrtf(ss) + bg[0];

    // softmax across the 32 lanes (all Q rows, no mask — matches reference)
    float mx = logit;
#pragma unroll
    for (int o = 16; o; o >>= 1) mx = fmaxf(mx, __shfl_xor(mx, o));
    float e = expf(logit - mx);
    float s = e;
#pragma unroll
    for (int o = 16; o; o >>= 1) s += __shfl_xor(s, o);
    const float g = e / s;

    // masked log1p histogram -> 30->5->1->1 tanh MLP
    const bool act = ln < qlen[b];
    const float* hp = hist + ((size_t)b * Q_SZ + ln) * NBINS;
    float h[NBINS];
#pragma unroll
    for (int i = 0; i < NBINS; ++i) h[i] = act ? log1pf(hp[i]) : 0.f;

    float z1[5];
#pragma unroll
    for (int j = 0; j < 5; ++j) {
        float a = b1[j];
        for (int i = 0; i < NBINS; ++i) a += h[i] * W1[j * NBINS + i];
        z1[j] = tanhf(a);
    }
    float a2 = b2[0];
#pragma unroll
    for (int j = 0; j < 5; ++j) a2 += z1[j] * W2[j];
    const float z2 = tanhf(a2);
    const float z3 = tanhf(b3[0] + z2 * W3[0]);

    float r = z3 * g;
#pragma unroll
    for (int o = 16; o; o >>= 1) r += __shfl_xor(r, o);
    if (ln == 0) out[b] = r;
}

// ---------------------------------------------------------------------------
extern "C" void kernel_launch(void* const* d_in, const int* in_sizes, int n_in,
                              void* d_out, int out_size, void* d_ws, size_t ws_size,
                              hipStream_t stream)
{
    const float* query = (const float*)d_in[0];
    const float* doc   = (const float*)d_in[1];
    const int*   qlen  = (const int*)  d_in[2];
    const float* W1    = (const float*)d_in[3];
    const float* b1    = (const float*)d_in[4];
    const float* W2    = (const float*)d_in[5];
    const float* b2    = (const float*)d_in[6];
    const float* W3    = (const float*)d_in[7];
    const float* b3    = (const float*)d_in[8];
    const float* Wg    = (const float*)d_in[9];
    const float* bg    = (const float*)d_in[10];
    float* out = (float*)d_out;

    // workspace layout: [ qn bf16 : B*Q*QSTR ][ hist f32 : B*Q*NBINS ]
    bf16*  qn   = (bf16*)d_ws;
    const size_t qn_bytes = (size_t)B_SZ * Q_SZ * QSTR * sizeof(bf16); // 1,343,488 (256-aligned)
    float* hist = (float*)((char*)d_ws + qn_bytes);

    drmm_qnorm<<<B_SZ * Q_SZ, 32, 0, stream>>>(query, qn, hist);
    drmm_interact_hist<<<B_SZ * NDT, 128, 0, stream>>>(doc, qn, hist);
    drmm_final<<<B_SZ, 32, 0, stream>>>(query, qlen, hist,
                                        W1, b1, W2, b2, W3, b3, Wg, bg, out);
}
